// BuildingBlockEmbedder_69870527971630
// MI455X (gfx1250) — matrix operand order repaired and checked by reading, not compiled, not run on hardware
//
#include <hip/hip_runtime.h>
#include <hip/hip_bf16.h>

// ---------------------------------------------------------------------------
// EGNN building-block embedder for MI455X (gfx1250, wave32, WMMA bf16).
// One workgroup == one building block (32 atoms, 640 intra-block edges).
// h[32x256] lives in LDS across all 4 layers; all GEMMs are
// v_wmma_f32_16x16x32_bf16 with fp32 accumulation.
// ---------------------------------------------------------------------------

typedef __attribute__((ext_vector_type(16))) __bf16        v16bf;
typedef __attribute__((ext_vector_type(8)))  float         v8f;
typedef __attribute__((ext_vector_type(4)))  unsigned int  v4u;

#define NPB   32
#define KNBR  20
#define EPB   (NPB * KNBR)   // 640 edges per block
#define CNODE 256
#define CHID  256
#define NLAY  4
#define NG    64             // gaussians
#define DEIN  577            // 2*256 + 1 + 64

__device__ __forceinline__ unsigned short f2bf(float x) {
  unsigned int u = __float_as_uint(x);
  u += 0x7FFFu + ((u >> 16) & 1u);            // round-to-nearest-even
  return (unsigned short)(u >> 16);
}

union BfPack { v16bf v; v4u q[2]; };

// A tile (16x32 bf16) from row-major LDS array: lane holds row M=lane&15,
// halves 0..7 = K kh..kh+7, halves 8..15 = K kh+16..kh+23, kh = (lane&16)?8:0.
__device__ __forceinline__ v16bf load_tile_a(const unsigned short* base, int stride,
                                             int rbase, int kbase, int lane) {
  int M  = lane & 15;
  int kh = kbase + ((lane & 16) ? 8 : 0);
  const unsigned short* p = base + (rbase + M) * stride + kh;
  BfPack u;
  u.q[0] = *reinterpret_cast<const v4u*>(p);        // ds/global b128
  u.q[1] = *reinterpret_cast<const v4u*>(p + 16);
  return u.v;
}

// B tile (32x16 bf16) from K-major transposed weights WT[n][k]:
// lane holds col n=nbase+(lane&15); halves 0..7 = K k0..k0+7, halves 8..15 = K k0+16..k0+23, k0=(lane&16)?kbase+8:kbase.
__device__ __forceinline__ v16bf load_tile_b(const unsigned short* wt, int kstride,
                                             int nbase, int kbase, int lane) {
  int n  = nbase + (lane & 15);
  int k0 = kbase + ((lane & 16) ? 8 : 0);
  const unsigned short* p = wt + n * kstride + k0;
  BfPack u;
  u.q[0] = *reinterpret_cast<const v4u*>(p);        // two 16-byte loads, 32 bytes apart
  u.q[1] = *reinterpret_cast<const v4u*>(p + 16);
  return u.v;
}

__device__ __forceinline__ v8f wmma_bf16(v16bf a, v16bf b, v8f c) {
  return __builtin_amdgcn_wmma_f32_16x16x32_bf16(false, a, false, b, (short)0, c,
                                                 false, false);
}

// ---------------------------------------------------------------------------
// Prep: W[L][K][256] fp32 -> out[L][256][ksub] bf16 (K-major, rows kstart..kstart+ksub)
// ---------------------------------------------------------------------------
__global__ void transpose_to_bf16(const float* __restrict__ W,
                                  unsigned short* __restrict__ out,
                                  int K, int kstart, int ksub) {
  int total = NLAY * 256 * ksub;
  for (int idx = blockIdx.x * blockDim.x + threadIdx.x; idx < total;
       idx += gridDim.x * blockDim.x) {
    int k   = idx % ksub;
    int rem = idx / ksub;
    int n   = rem & 255;
    int l   = rem >> 8;
    out[idx] = f2bf(W[(l * K + kstart + k) * 256 + n]);
  }
}

// ---------------------------------------------------------------------------
// Main fused kernel: one block per building block, 256 threads = 8 wave32.
// Wave w owns output columns [32w, 32w+32).
// ---------------------------------------------------------------------------
__global__ void __launch_bounds__(256)
egnn_block_forward(const float* __restrict__ pos,          // [N,3] Angstrom
                   const int* __restrict__ atype,          // [N] 1..100
                   const int* __restrict__ ecol,           // [E] global col ids
                   const float* __restrict__ embed,        // [100,256]
                   const float* __restrict__ offs,         // [64]
                   const float* __restrict__ We1f,         // [L,577,256] (for w_r row 512)
                   const float* __restrict__ be1,
                   const float* __restrict__ be2,
                   const float* __restrict__ bn1,
                   const float* __restrict__ bn2,
                   const unsigned short* __restrict__ We1aT, // [L,256,256]
                   const unsigned short* __restrict__ We1bT, // [L,256,256]
                   const unsigned short* __restrict__ We1cT, // [L,256,64]
                   const unsigned short* __restrict__ We2T,  // [L,256,256]
                   const unsigned short* __restrict__ Wn1T,  // [L,256,512]
                   const unsigned short* __restrict__ Wn2T,  // [L,256,256]
                   float* __restrict__ out)                  // [B,256]
{
  extern __shared__ char smem[];
  float* hf   = (float*)smem;            // [32][256] fp32 master h
  float* aggf = hf   + NPB * CNODE;      // [32][256]
  float* Pf   = aggf + NPB * CNODE;      // [32][256]  h @ We1a
  float* Qf   = Pf   + NPB * CNODE;      // [32][256]  h @ We1b
  float* rad  = Qf   + NPB * CNODE;      // [640]
  int*   coln = (int*)(rad + EPB);       // [640] local col index
  unsigned short* hb = (unsigned short*)(coln + EPB); // [32][256] bf16 h
  unsigned short* ab = hb + NPB * CNODE;              // [32][256] bf16 agg
  unsigned short* t1 = ab + NPB * CNODE;              // [32][256] node hidden
  unsigned short* mt = t1 + NPB * CNODE;              // [16][256] edge msg tile
  unsigned short* ea = mt + 16 * CNODE;               // [640][64] edge_attr bf16

  const int tid   = threadIdx.x;
  const int lane  = tid & 31;
  const int w     = tid >> 5;                 // wave id 0..7
  const int b     = blockIdx.x;
  const int abase = b * NPB;

  const float GC = -0.5f / ((5.0f / 63.0f) * (5.0f / 63.0f));  // gauss coeff

  // ----- Phase 0: load h, edge geometry, gaussian smearing (layer-invariant)
  for (int i = tid; i < NPB * CNODE; i += 256) {
    int a = i >> 8, c = i & 255;
    int t = atype[abase + a];
    hf[i] = embed[(t - 1) * CNODE + c];
  }
  for (int e = tid; e < EPB; e += 256) {
    int r  = e / KNBR;
    int cl = ecol[b * EPB + e] - abase;
    coln[e] = cl;
    float rx = pos[(abase + r) * 3 + 0], ry = pos[(abase + r) * 3 + 1],
          rz = pos[(abase + r) * 3 + 2];
    float cx = pos[(abase + cl) * 3 + 0], cy = pos[(abase + cl) * 3 + 1],
          cz = pos[(abase + cl) * 3 + 2];
    float dx = cx - rx, dy = cy - ry, dz = cz - rz;
    float d2 = dx * dx + dy * dy + dz * dz;
    rad[e]   = 0.01f * d2;                    // (0.1*d)^2
    float dist = sqrtf(d2);
    for (int j = 0; j < NG; ++j) {
      float t0 = dist - offs[j];
      ea[e * NG + j] = f2bf(__expf(GC * t0 * t0));
    }
  }
  __syncthreads();

  for (int l = 0; l < NLAY; ++l) {
    const unsigned short* wA  = We1aT + l * 256 * 256;
    const unsigned short* wB  = We1bT + l * 256 * 256;
    const unsigned short* wC  = We1cT + l * 256 * 64;
    const unsigned short* w2  = We2T  + l * 256 * 256;
    const unsigned short* wn1 = Wn1T  + l * 256 * 512;
    const unsigned short* wn2 = Wn2T  + l * 256 * 256;
    const float* b1  = be1 + l * 256;
    const float* b2  = be2 + l * 256;
    const float* bb1 = bn1 + l * 256;
    const float* bb2 = bn2 + l * 256;
    const float* wrp = We1f + (l * DEIN + 512) * 256;   // radial weight row

    // ----- 1. h -> bf16, zero agg
    for (int i = tid; i < NPB * CNODE; i += 256) {
      hb[i]   = f2bf(hf[i]);
      aggf[i] = 0.f;
    }
    __syncthreads();

    // ----- 2. P = h@We1a, Q = h@We1b  (M=32, K=256, N=256)
    for (int rt = 0; rt < 2; ++rt) {
      for (int ct = 0; ct < 2; ++ct) {
        int nb = w * 32 + ct * 16;
        v8f accP = {}, accQ = {};
#pragma unroll
        for (int kt = 0; kt < 8; ++kt) {
          v16bf A  = load_tile_a(hb, 256, rt * 16, kt * 32, lane);
          v16bf Ba = load_tile_b(wA, 256, nb, kt * 32, lane);
          v16bf Bb = load_tile_b(wB, 256, nb, kt * 32, lane);
          accP = wmma_bf16(A, Ba, accP);
          accQ = wmma_bf16(A, Bb, accQ);
        }
#pragma unroll
        for (int v = 0; v < 8; ++v) {
          int M = rt * 16 + v + ((lane >> 4) << 3);
          int c = nb + (lane & 15);
          Pf[M * 256 + c] = accP[v];
          Qf[M * 256 + c] = accQ[v];
        }
      }
    }
    __syncthreads();

    // ----- 3. Edge MLP over 40 tiles of 16 edges
    for (int et = 0; et < 40; ++et) {
      int ebase = et * 16;
      // 3a: m = relu(P[row] + Q[col] + radial*wr + ea@We1c + be1)
      for (int ct = 0; ct < 2; ++ct) {
        int nb = w * 32 + ct * 16;
        int c  = nb + (lane & 15);
        float wrc = wrp[c];
        float b1c = b1[c];
        v8f acc = {};
#pragma unroll
        for (int kt = 0; kt < 2; ++kt) {
          v16bf A  = load_tile_a(ea, NG, ebase, kt * 32, lane);
          v16bf Bx = load_tile_b(wC, 64, nb, kt * 32, lane);
          acc = wmma_bf16(A, Bx, acc);
        }
#pragma unroll
        for (int v = 0; v < 8; ++v) {
          int M  = v + ((lane >> 4) << 3);
          int e  = ebase + M;
          int r  = e / KNBR;
          int cl = coln[e];
          float val = acc[v] + Pf[r * 256 + c] + Qf[cl * 256 + c] +
                      rad[e] * wrc + b1c;
          mt[M * 256 + c] = f2bf(fmaxf(val, 0.f));
        }
      }
      __syncthreads();
      // 3b: m2 = relu(m@We2 + be2); agg[row] += m2
      for (int ct = 0; ct < 2; ++ct) {
        int nb = w * 32 + ct * 16;
        int c  = nb + (lane & 15);
        float b2c = b2[c];
        v8f acc = {};
#pragma unroll
        for (int kt = 0; kt < 8; ++kt) {
          v16bf A  = load_tile_a(mt, 256, 0, kt * 32, lane);
          v16bf Bx = load_tile_b(w2, 256, nb, kt * 32, lane);
          acc = wmma_bf16(A, Bx, acc);
        }
#pragma unroll
        for (int v = 0; v < 8; ++v) {
          int M = v + ((lane >> 4) << 3);
          int e = ebase + M;
          int r = e / KNBR;
          float val = fmaxf(acc[v] + b2c, 0.f);
          atomicAdd(&aggf[r * 256 + c], val);     // ds_add_f32 (wave-owned col)
        }
      }
      __syncthreads();
    }

    // ----- 4. Node MLP: t1 = relu([h|agg]@Wn1+bn1); h = 2h + t1@Wn2 + bn2
    for (int i = tid; i < NPB * CNODE; i += 256) ab[i] = f2bf(aggf[i]);
    __syncthreads();

    for (int rt = 0; rt < 2; ++rt) {
      for (int ct = 0; ct < 2; ++ct) {
        int nb = w * 32 + ct * 16;
        int c  = nb + (lane & 15);
        v8f acc = {};
#pragma unroll
        for (int kt = 0; kt < 8; ++kt) {          // K = 0..255 : h part
          v16bf A  = load_tile_a(hb, 256, rt * 16, kt * 32, lane);
          v16bf Bx = load_tile_b(wn1, 512, nb, kt * 32, lane);
          acc = wmma_bf16(A, Bx, acc);
        }
#pragma unroll
        for (int kt = 0; kt < 8; ++kt) {          // K = 256..511 : agg part
          v16bf A  = load_tile_a(ab, 256, rt * 16, kt * 32, lane);
          v16bf Bx = load_tile_b(wn1, 512, nb, 256 + kt * 32, lane);
          acc = wmma_bf16(A, Bx, acc);
        }
        float bc = bb1[c];
#pragma unroll
        for (int v = 0; v < 8; ++v) {
          int M = rt * 16 + v + ((lane >> 4) << 3);
          t1[M * 256 + c] = f2bf(fmaxf(acc[v] + bc, 0.f));
        }
      }
    }
    __syncthreads();

    for (int rt = 0; rt < 2; ++rt) {
      for (int ct = 0; ct < 2; ++ct) {
        int nb = w * 32 + ct * 16;
        int c  = nb + (lane & 15);
        v8f acc = {};
#pragma unroll
        for (int kt = 0; kt < 8; ++kt) {
          v16bf A  = load_tile_a(t1, 256, rt * 16, kt * 32, lane);
          v16bf Bx = load_tile_b(wn2, 256, nb, kt * 32, lane);
          acc = wmma_bf16(A, Bx, acc);
        }
        float bc = bb2[c];
#pragma unroll
        for (int v = 0; v < 8; ++v) {
          int M   = rt * 16 + v + ((lane >> 4) << 3);
          int idx = M * 256 + c;
          hf[idx] = 2.f * hf[idx] + acc[v] + bc;  // h + (h + out)
        }
      }
    }
    __syncthreads();
  }

  // ----- mean pool over the 32 atoms of the block
  float s = 0.f;
  for (int a = 0; a < NPB; ++a) s += hf[a * 256 + tid];
  out[b * 256 + tid] = s * (1.0f / NPB);
}

// ---------------------------------------------------------------------------
extern "C" void kernel_launch(void* const* d_in, const int* in_sizes, int n_in,
                              void* d_out, int out_size, void* d_ws, size_t ws_size,
                              hipStream_t stream) {
  const float* pos    = (const float*)d_in[0];
  const int*   atype  = (const int*)  d_in[1];
  const int*   eidx   = (const int*)  d_in[2];
  const float* embed  = (const float*)d_in[4];
  const float* offs   = (const float*)d_in[5];
  const float* We1    = (const float*)d_in[6];
  const float* be1    = (const float*)d_in[7];
  const float* We2    = (const float*)d_in[8];
  const float* be2    = (const float*)d_in[9];
  const float* Wn1    = (const float*)d_in[10];
  const float* bn1    = (const float*)d_in[11];
  const float* Wn2    = (const float*)d_in[12];
  const float* bn2    = (const float*)d_in[13];
  float* outp = (float*)d_out;

  const int N = in_sizes[1];       // 16384 atoms
  const int B = N / NPB;           // 512 blocks
  const int E = in_sizes[2] / 2;   // 327680 edges

  // Workspace: transposed bf16 weights (~3.3 MB total)
  unsigned short* ws = (unsigned short*)d_ws;
  unsigned short* We1aT = ws;                       // [L,256,256]
  unsigned short* We1bT = We1aT + NLAY * 256 * 256;
  unsigned short* We1cT = We1bT + NLAY * 256 * 256; // [L,256,64]
  unsigned short* We2T  = We1cT + NLAY * 256 * 64;
  unsigned short* Wn1T  = We2T  + NLAY * 256 * 256; // [L,256,512]
  unsigned short* Wn2T  = Wn1T  + NLAY * 256 * 512;

  transpose_to_bf16<<<512, 256, 0, stream>>>(We1, We1aT, DEIN,   0, 256);
  transpose_to_bf16<<<512, 256, 0, stream>>>(We1, We1bT, DEIN, 256, 256);
  transpose_to_bf16<<<512, 256, 0, stream>>>(We1, We1cT, DEIN, 513,  64);
  transpose_to_bf16<<<512, 256, 0, stream>>>(We2, We2T,   256,   0, 256);
  transpose_to_bf16<<<512, 256, 0, stream>>>(Wn1, Wn1T,   512,   0, 512);
  transpose_to_bf16<<<512, 256, 0, stream>>>(Wn2, Wn2T,   256,   0, 256);

  // Dynamic LDS: 4 fp32 [32][256] + rad + coln + 3 bf16 [32][256] + mt + ea
  const size_t smem =
      (size_t)4 * NPB * CNODE * 4 + EPB * 4 + EPB * 4 +     // fp32 + int
      (size_t)3 * NPB * CNODE * 2 + 16 * CNODE * 2 + EPB * NG * 2;  // bf16

  egnn_block_forward<<<B, 256, smem, stream>>>(
      pos, atype, eidx + E, embed, offs, We1, be1, be2, bn1, bn2,
      We1aT, We1bT, We1cT, We2T, Wn1T, Wn2T, outp);

  (void)n_in; (void)out_size; (void)ws_size;
}
